// Net_SLSTM_Conv_56272661512583
// MI455X (gfx1250) — matrix operand order repaired and checked by reading, not compiled, not run on hardware
//
#include <hip/hip_runtime.h>
#include <hip/hip_bf16.h>

typedef __attribute__((ext_vector_type(16))) _Float16 v16h;
typedef __attribute__((ext_vector_type(8)))  _Float16 v8h;
typedef __attribute__((ext_vector_type(8)))  float    v8f;

// ---------------------------------------------------------------------------
// Fast math helpers (v_exp_f32 transcendental path)
// ---------------------------------------------------------------------------
__device__ __forceinline__ float fsig(float x) {
    return 1.0f / (1.0f + __expf(-x));
}
__device__ __forceinline__ float ftanh(float x) {
    float e = __expf(-2.0f * x);
    return (1.0f - e) / (1.0f + e);
}

// ---------------------------------------------------------------------------
// WMMA fragment helpers (layouts per CDNA5 ISA §7.12.2, wave32)
// ---------------------------------------------------------------------------
__device__ __forceinline__ v16h frag_combine(v8h lo, v8h hi) {
    v16h a;
#pragma unroll
    for (int i = 0; i < 8; ++i) { a[i] = lo[i]; a[8 + i] = hi[i]; }
    return a;
}

// A-matrix 16x32 f16 fragment. base -> element (row 0, k0); rowstride in halves.
// Lane l (m = l&15, g = l>>4) holds K chunks [g*8, g*8+8) and [16+g*8, 16+g*8+8).
__device__ __forceinline__ v16h load_a_frag(const _Float16* base, int rowstride, int lane) {
    const int m = lane & 15, g = lane >> 4;
    const _Float16* p = base + (size_t)m * rowstride + g * 8;
    v8h lo = *(const v8h*)(p);
    v8h hi = *(const v8h*)(p + 16);
    return frag_combine(lo, hi);
}

// Same, but multiplies each K element by a per-K fp16 scale (BN fold for layer 2).
__device__ __forceinline__ v16h load_a_frag_scaled(const _Float16* base, int rowstride,
                                                   const _Float16* scal, int lane) {
    const int m = lane & 15, g = lane >> 4;
    const _Float16* p = base + (size_t)m * rowstride + g * 8;
    v8h lo = *(const v8h*)(p);
    v8h hi = *(const v8h*)(p + 16);
    v8h sl = *(const v8h*)(scal + g * 8);
    v8h sh = *(const v8h*)(scal + 16 + g * 8);
    lo *= sl;
    hi *= sh;
    return frag_combine(lo, hi);
}

// B-matrix 32x16 f16 fragment from LDS weights stored [K][512] row-major.
// Lane l (n = l&15, g = l>>4) holds column n, K rows [g*8, g*8+8) and [16+g*8, 16+g*8+8).
__device__ __forceinline__ v16h load_b_frag(const _Float16* base, int lane) {
    const int n = lane & 15, g = lane >> 4;
    v16h b;
#pragma unroll
    for (int i = 0; i < 8; ++i) {
        b[i]     = base[(size_t)(g * 8 + i) * 512 + n];
        b[8 + i] = base[(size_t)(16 + g * 8 + i) * 512 + n];
    }
    return b;
}

__device__ __forceinline__ v8f wmma_acc(v16h a, v16h b, v8f c) {
    return __builtin_amdgcn_wmma_f32_16x16x32_f16(false, a, false, b, (short)0, c,
                                                  false, false);
}

// ---------------------------------------------------------------------------
// Kernel 1: conv1d over time (kernel 3, SAME) + ATan-spike -> fp16 spikes
// x: [B=256][T=512][C=14] f32;  cur1: [B][T][32] f16 in {0,1}
// ---------------------------------------------------------------------------
__global__ __launch_bounds__(256) void conv_spike_kernel(
    const float* __restrict__ x, const float* __restrict__ conv_w,
    const float* __restrict__ conv_b, _Float16* __restrict__ cur1) {
    constexpr int T = 512, C = 14, OC = 32;
    __shared__ float w[OC * C * 3];
    __shared__ float b[OC];
    for (int i = threadIdx.x; i < OC * C * 3; i += blockDim.x) w[i] = conv_w[i];
    if (threadIdx.x < OC) b[threadIdx.x] = conv_b[threadIdx.x];
    __syncthreads();

    int idx = blockIdx.x * blockDim.x + threadIdx.x;   // over B*T
    if (idx >= 256 * T) return;
    int t = idx & (T - 1);
    const float* xr = x + (size_t)idx * C;

    float xm[C], x0[C], xp[C];
#pragma unroll
    for (int c = 0; c < C; ++c) {
        x0[c] = xr[c];
        xm[c] = (t > 0)     ? xr[c - C] : 0.0f;
        xp[c] = (t < T - 1) ? xr[c + C] : 0.0f;
    }
    _Float16 outv[OC];
#pragma unroll
    for (int oc = 0; oc < OC; ++oc) {
        const float* wr = w + oc * C * 3;
        float acc = b[oc];
#pragma unroll
        for (int c = 0; c < C; ++c)
            acc += wr[c * 3 + 0] * xm[c] + wr[c * 3 + 1] * x0[c] + wr[c * 3 + 2] * xp[c];
        outv[oc] = (acc - 1.0f > 0.0f) ? (_Float16)1.0f : (_Float16)0.0f;
    }
    v8h* dst = (v8h*)(cur1 + (size_t)idx * OC);
#pragma unroll
    for (int j = 0; j < 4; ++j) {
        v8h v;
#pragma unroll
        for (int i = 0; i < 8; ++i) v[i] = outv[j * 8 + i];
        dst[j] = v;
    }
}

// ---------------------------------------------------------------------------
// Kernel 2/5: SLSTM scan layer. One workgroup owns 16 rows of the Teff=512
// batch and runs all 256 sequential steps with state + fp16 weights in LDS.
// Per step, wave w computes gate tiles {w, w+8, w+16, w+24} (i,f,g,o for
// h in [16w,16w+16)) via v_wmma_f32_16x16x32_f16.
// ---------------------------------------------------------------------------
template <int DIN, bool LAYER2>
__global__ __launch_bounds__(256) void slstm_layer_kernel(
    const _Float16* __restrict__ cur,   // [256][512][DIN] f16
    const float* __restrict__ w_ih,     // [512][DIN]
    const float* __restrict__ w_hh,     // [512][128]
    const float* __restrict__ b_ih,     // [512]
    const float* __restrict__ b_hh,     // [512]
    const float* __restrict__ thr_p,    // scalar
    const _Float16* __restrict__ a_h,   // [128] BN scale (LAYER2)
    const float* __restrict__ cvec,     // [128] BN shift (LAYER2)
    _Float16* __restrict__ spk_out,     // [256][512][128] f16 (!LAYER2)
    float* __restrict__ memsum_out)     // [512][128] f32 (LAYER2)
{
    constexpr int H = 128, NG = 512, TEFF = 512, STEPS = 256, ROWS = 16;
    extern __shared__ char smem[];
    _Float16* lds_wih  = (_Float16*)smem;            // [DIN][512]
    _Float16* lds_whh  = lds_wih + DIN * NG;         // [128][512]
    _Float16* lds_memh = lds_whh + H * NG;           // [16][128] fp16 state copy
    _Float16* lds_a    = lds_memh + ROWS * H;        // [128] BN scale
    float* lds_bias = (float*)(lds_a + H);           // [512]
    float* lds_mem  = lds_bias + NG;                 // [16][128]
    float* lds_syn  = lds_mem + ROWS * H;            // [16][128]
    float* lds_sum  = lds_syn + ROWS * H;            // [16][128] (LAYER2 only)

    const int tid  = threadIdx.x;
    const int lane = tid & 31;
    const int wv   = tid >> 5;                       // 0..7 (8 wave32 waves)
    const int t0   = blockIdx.x * ROWS;
    const float thr = *thr_p;

    // Stage weights to LDS transposed to [K][N] fp16 (B-fragment friendly).
    for (int i = tid; i < DIN * NG; i += 256) {
        int k = i / NG, n = i % NG;
        lds_wih[i] = (_Float16)w_ih[(size_t)n * DIN + k];
    }
    for (int i = tid; i < H * NG; i += 256) {
        int k = i / NG, n = i % NG;
        lds_whh[i] = (_Float16)w_hh[(size_t)n * H + k];
    }
    // Gate bias; layer2 folds the BN shift through w_ih: + cvec . w_ih[n,:]
    for (int n = tid; n < NG; n += 256) {
        float bb = b_ih[n] + b_hh[n];
        if (LAYER2) {
            const float* wr = w_ih + (size_t)n * DIN;
            for (int k = 0; k < H; ++k) bb += cvec[k] * wr[k];
        }
        lds_bias[n] = bb;
    }
    for (int i = tid; i < ROWS * H; i += 256) {
        lds_mem[i] = 0.0f;
        lds_syn[i] = 0.0f;
        lds_memh[i] = (_Float16)0.0f;
        if (LAYER2) lds_sum[i] = 0.0f;
    }
    if (LAYER2 && tid < H) lds_a[tid] = a_h[tid];
    __syncthreads();

    const int hcol = wv * 16 + (lane & 15);   // D-frag column this lane owns
    const int g16  = lane >> 4;               // D-frag row group

    for (int step = 0; step < STEPS; ++step) {
        const _Float16* curbase = cur + ((size_t)step * TEFF + t0) * DIN;

        // A fragments: input (global, optionally BN-scaled) + recurrent (LDS)
        v16h a_in[DIN / 32];
#pragma unroll
        for (int kt = 0; kt < DIN / 32; ++kt) {
            if (LAYER2)
                a_in[kt] = load_a_frag_scaled(curbase + kt * 32, DIN, lds_a + kt * 32, lane);
            else
                a_in[kt] = load_a_frag(curbase + kt * 32, DIN, lane);
        }
        v16h a_m[4];
#pragma unroll
        for (int kt = 0; kt < 4; ++kt)
            a_m[kt] = load_a_frag(lds_memh + kt * 32, H, lane);
        __syncthreads();   // all waves done reading lds_memh before updates

        if (step + 1 < STEPS)
            __builtin_prefetch(curbase + (size_t)TEFF * DIN, 0, 0);

        // Gate GEMMs: i,f,g,o tiles for this wave's h-range
        v8f gacc[4];
#pragma unroll
        for (int gg = 0; gg < 4; ++gg) {
            const int n0 = (wv + 8 * gg) * 16;
            const float bv = lds_bias[n0 + (lane & 15)];
            v8f acc;
#pragma unroll
            for (int r = 0; r < 8; ++r) acc[r] = bv;
#pragma unroll
            for (int kt = 0; kt < DIN / 32; ++kt) {
                v16h bfr = load_b_frag(lds_wih + (size_t)kt * 32 * NG + n0, lane);
                acc = wmma_acc(a_in[kt], bfr, acc);
            }
#pragma unroll
            for (int kt = 0; kt < 4; ++kt) {
                v16h bfr = load_b_frag(lds_whh + (size_t)kt * 32 * NG + n0, lane);
                acc = wmma_acc(a_m[kt], bfr, acc);
            }
            gacc[gg] = acc;
        }

        // LSTM cell update for (rows r+8*g16, column hcol)
#pragma unroll
        for (int r = 0; r < 8; ++r) {
            const int m = r + 8 * g16;
            const int sidx = m * H + hcol;
            float ig  = fsig(gacc[0][r]);
            float fg  = fsig(gacc[1][r]);
            float gg_ = ftanh(gacc[2][r]);
            float og  = fsig(gacc[3][r]);
            float syn_old = lds_syn[sidx];
            float mem_old = lds_mem[sidx];
            float syn_new = fg * syn_old + ig * gg_;
            float reset   = (mem_old - thr > 0.0f) ? thr : 0.0f;  // from PREVIOUS mem
            float mem_new = og * ftanh(syn_new) - reset;
            lds_syn[sidx]  = syn_new;
            lds_mem[sidx]  = mem_new;
            lds_memh[sidx] = (_Float16)mem_new;
            if (!LAYER2) {
                _Float16 spk = (mem_new - thr > 0.0f) ? (_Float16)1.0f : (_Float16)0.0f;
                spk_out[((size_t)step * TEFF + (t0 + m)) * H + hcol] = spk;
            } else {
                lds_sum[sidx] += mem_new;
            }
        }
        __syncthreads();   // state visible before next step's fragment loads
    }

    if (LAYER2) {
        for (int i = tid; i < ROWS * H; i += 256) {
            int m = i / H, h = i % H;
            memsum_out[(size_t)(t0 + m) * H + h] = lds_sum[i] * (1.0f / 256.0f);
        }
    }
}

// ---------------------------------------------------------------------------
// Kernel 3a: zero the BN count accumulator
// ---------------------------------------------------------------------------
__global__ void zero_kernel(float* p, int n) {
    int i = blockIdx.x * blockDim.x + threadIdx.x;
    if (i < n) p[i] = 0.0f;
}

// ---------------------------------------------------------------------------
// Kernel 3b: per-channel spike counts (spikes are binary: mean determines var)
// ---------------------------------------------------------------------------
__global__ __launch_bounds__(256) void bn_count_kernel(
    const _Float16* __restrict__ spk, float* __restrict__ counts, int total8) {
    __shared__ float bins[128];
    if (threadIdx.x < 128) bins[threadIdx.x] = 0.0f;
    __syncthreads();
    int idx = blockIdx.x * blockDim.x + threadIdx.x;   // each handles 8 halves
    if (idx < total8) {
        v8h v = *(const v8h*)(spk + (size_t)idx * 8);
        int hbase = (idx * 8) & 127;
#pragma unroll
        for (int i = 0; i < 8; ++i) atomicAdd(&bins[hbase + i], (float)v[i]);
    }
    __syncthreads();
    if (threadIdx.x < 128) atomicAdd(&counts[threadIdx.x], bins[threadIdx.x]);
}

// ---------------------------------------------------------------------------
// Kernel 4: BN prep.  mu = count/N; var = mu*(1-mu) for binary inputs.
// a = gamma*rsqrt(var+eps) (fp16 scale), cvec = beta - mu*a (fp32 shift).
// ---------------------------------------------------------------------------
__global__ void bn_prep_kernel(const float* __restrict__ counts,
                               const float* __restrict__ gamma,
                               const float* __restrict__ beta,
                               _Float16* __restrict__ a_h,
                               float* __restrict__ cvec) {
    int h = threadIdx.x;
    if (h < 128) {
        float mu  = counts[h] * (1.0f / 131072.0f);
        float var = mu - mu * mu;
        float a   = gamma[h] * rsqrtf(var + 1e-5f);
        a_h[h]  = (_Float16)a;
        cvec[h] = beta[h] - mu * a;
    }
}

// ---------------------------------------------------------------------------
// Kernel 6: final FC  out[t][nc] = final_mem[t,:] . fc_w[nc,:] + fc_b[nc]
// ---------------------------------------------------------------------------
__global__ __launch_bounds__(256) void fc_kernel(const float* __restrict__ fm,
                                                 const float* __restrict__ fw,
                                                 const float* __restrict__ fb,
                                                 float* __restrict__ out) {
    int id = blockIdx.x * blockDim.x + threadIdx.x;
    if (id >= 512 * 8) return;
    int t = id >> 3, nc = id & 7;
    const float* f = fm + (size_t)t * 128;
    const float* w = fw + (size_t)nc * 128;
    float acc = fb[nc];
#pragma unroll 8
    for (int h = 0; h < 128; ++h) acc += f[h] * w[h];
    out[id] = acc;
}

// ---------------------------------------------------------------------------
// Host launcher
// ---------------------------------------------------------------------------
extern "C" void kernel_launch(void* const* d_in, const int* in_sizes, int n_in,
                              void* d_out, int out_size, void* d_ws, size_t ws_size,
                              hipStream_t stream) {
    const float* x      = (const float*)d_in[0];
    const float* conv_w = (const float*)d_in[1];
    const float* conv_b = (const float*)d_in[2];
    const float* w_ih1  = (const float*)d_in[3];
    const float* w_hh1  = (const float*)d_in[4];
    const float* b_ih1  = (const float*)d_in[5];
    const float* b_hh1  = (const float*)d_in[6];
    const float* thr1   = (const float*)d_in[7];
    const float* w_ih2  = (const float*)d_in[8];
    const float* w_hh2  = (const float*)d_in[9];
    const float* b_ih2  = (const float*)d_in[10];
    const float* b_hh2  = (const float*)d_in[11];
    const float* thr2   = (const float*)d_in[12];
    const float* bn_g   = (const float*)d_in[13];
    const float* bn_b   = (const float*)d_in[14];
    const float* fc_w   = (const float*)d_in[15];
    const float* fc_b   = (const float*)d_in[16];
    float* out = (float*)d_out;

    // Workspace carve
    char* ws = (char*)d_ws;
    _Float16* cur1 = (_Float16*)ws;                               // 8 MiB
    _Float16* spk1 = (_Float16*)(ws + 8388608);                   // 32 MiB
    float*    counts = (float*)(ws + 8388608 + 33554432);         // 512 B
    _Float16* a_h    = (_Float16*)(ws + 41943040 + 512);          // 256 B
    float*    cvec   = (float*)(ws + 41943040 + 1024);            // 512 B
    float*    finalm = (float*)(ws + 41943040 + 2048);            // 256 KiB

    // 1) conv + spike
    conv_spike_kernel<<<512, 256, 0, stream>>>(x, conv_w, conv_b, cur1);

    // zero BN counts (before layer1 finishes is fine: same stream order)
    zero_kernel<<<1, 128, 0, stream>>>(counts, 128);

    // 2) SLSTM layer 1 (Din=32): 32 workgroups x 16 rows, 256 steps
    // LDS: wih(32KB f16) + whh(128KB f16) + state/bias = 186,624 B
    slstm_layer_kernel<32, false><<<32, 256, 186624, stream>>>(
        cur1, w_ih1, w_hh1, b_ih1, b_hh1, thr1, nullptr, nullptr, spk1, nullptr);

    // 3) BN stats: per-channel counts of binary spikes
    bn_count_kernel<<<8192, 256, 0, stream>>>(spk1, counts, 131072 * 128 / 8);

    // 4) BN fold coefficients
    bn_prep_kernel<<<1, 128, 0, stream>>>(counts, bn_g, bn_b, a_h, cvec);

    // 5) SLSTM layer 2 (Din=128, BN folded): LDS = 293,120 B (< 320KB/WGP)
    slstm_layer_kernel<128, true><<<32, 256, 293120, stream>>>(
        spk1, w_ih2, w_hh2, b_ih2, b_hh2, thr2, a_h, cvec, nullptr, finalm);

    // 6) FC head [512,128] @ [128,8]
    fc_kernel<<<16, 256, 0, stream>>>(finalm, fc_w, fc_b, out);
}
